// Attn_66297115181215
// MI455X (gfx1250) — matrix operand order repaired and checked by reading, not compiled, not run on hardware
//
#include <hip/hip_runtime.h>

typedef __bf16 bf16;
typedef __attribute__((ext_vector_type(8)))  __bf16 bf16x8;
typedef __attribute__((ext_vector_type(16))) __bf16 bf16x16;
typedef __attribute__((ext_vector_type(8)))  float  f32x8;

// Padded LDS row length for 32-deep K tiles: 40 elems = 80B (16B-aligned rows).
#define LDK 40

// ---- CDNA5 async global->LDS copy (ASYNCcnt path), guarded ----------------
// Probe round revealed param type: 'vector_size(4*sizeof(int)) int __device__ *'
// i.e. v4i in addrspace(1); LDS side analogously v4i in addrspace(3).
#if __has_builtin(__builtin_amdgcn_global_load_async_to_lds_b128)
#define HAVE_ASYNC 1
typedef int v4i __attribute__((vector_size(16)));
#define ASYNC_CP16(lds_dst, g_src)                                             \
    __builtin_amdgcn_global_load_async_to_lds_b128(                            \
        (__attribute__((address_space(1))) v4i*)(g_src),                       \
        (__attribute__((address_space(3))) v4i*)(lds_dst), 0, 0)
__device__ __forceinline__ void wait_async0() {
#if __has_builtin(__builtin_amdgcn_s_wait_asynccnt)
    __builtin_amdgcn_s_wait_asynccnt(0);
#else
    asm volatile("s_wait_asynccnt 0" ::: "memory");
#endif
}
#else
#define HAVE_ASYNC 0
__device__ __forceinline__ void wait_async0() {}
#endif

// ---- WMMA fragment loaders (bf16, 16x16x32) -------------------------------
// A (MxK=16x32): lane m=lane&15; lanes 0-15 hold K {0..7, 16..23},
//                lanes 16-31 hold K {8..15, 24..31}  -> chunks at +0/+16 (lhi adds +8)
__device__ __forceinline__ bf16x16 ld_fragA(const bf16* row, int lhi) {
    const bf16* p = row + (lhi ? 8 : 0);
    bf16x8 lo = *(const bf16x8*)(p);
    bf16x8 up = *(const bf16x8*)(p + 16);
    return __builtin_shufflevector(lo, up, 0,1,2,3,4,5,6,7,8,9,10,11,12,13,14,15);
}
// B (KxN=32x16) staged transposed as [n][k]: lane n=lane&15; lanes 0-15 hold
// K {0..7, 16..23}, lanes 16-31 hold K {8..15, 24..31} -> chunks at +0/+16 (lhi adds +8)
__device__ __forceinline__ bf16x16 ld_fragB(const bf16* row, int lhi) {
    const bf16* p = row + (lhi ? 8 : 0);
    bf16x8 lo = *(const bf16x8*)(p);
    bf16x8 up = *(const bf16x8*)(p + 16);
    return __builtin_shufflevector(lo, up, 0,1,2,3,4,5,6,7,8,9,10,11,12,13,14,15);
}

#define WMMA_BF16(ACC, A, Bf) \
    (ACC) = __builtin_amdgcn_wmma_f32_16x16x32_bf16(false, (A), false, (Bf), (short)0, (ACC), false, false)

// Problem constants
#define SB 2
#define SN 8192
#define SC 256
#define SH 8
#define SCH 2048          // C*H
#define SM 16384          // B*N

// ===========================================================================
// Kernel 1: fused QKV projection.  P = X[16384,256] x W[256,2048] -> bf16.
// grid (16, 256, 3); block tile 64(M) x 128(N); wave tile 32x32 (2x2 WMMA).
// ===========================================================================
__global__ void qkv_gemm(const float* __restrict__ X,
                         const float* __restrict__ Wq,
                         const float* __restrict__ Wk,
                         const float* __restrict__ Wv,
                         bf16* __restrict__ Qb, bf16* __restrict__ Kb,
                         bf16* __restrict__ Vb)
{
    const float* W = (blockIdx.z == 0) ? Wq : (blockIdx.z == 1) ? Wk : Wv;
    bf16*        P = (blockIdx.z == 0) ? Qb : (blockIdx.z == 1) ? Kb : Vb;
    const int m0 = blockIdx.y * 64;
    const int n0 = blockIdx.x * 128;

    __shared__ __align__(16) bf16 sA[64 * LDK];
    __shared__ __align__(16) bf16 sB[128 * LDK];

    const int tid  = threadIdx.x;
    const int wave = tid >> 5, lane = tid & 31;
    const int lrow = lane & 15, lhi = lane >> 4;
    const int wm = (wave >> 2) * 32;   // 0 / 32
    const int wn = (wave & 3) * 32;    // 0..96

    f32x8 acc[2][2] = {};

    for (int kt = 0; kt < SC; kt += 32) {
        {   // stage A: X tile, fp32 -> bf16, float4 coalesced
            const int c4 = tid & 7;       // float4 column
            const int r  = tid >> 3;      // 0..31
#pragma unroll
            for (int rr = r; rr < 64; rr += 32) {
                float4 v = *(const float4*)(X + (m0 + rr) * SC + kt + c4 * 4);
                bf16* d = sA + rr * LDK + c4 * 4;
                d[0] = (bf16)v.x; d[1] = (bf16)v.y; d[2] = (bf16)v.z; d[3] = (bf16)v.w;
            }
        }
        {   // stage B transposed: sB[n][k] = W[kt+k][n0+n]
            const int n  = tid & 127;
            const int k0 = tid >> 7;      // 0..1
#pragma unroll
            for (int kk = k0; kk < 32; kk += 2)
                sB[n * LDK + kk] = (bf16)W[(kt + kk) * SCH + n0 + n];
        }
        __syncthreads();
#pragma unroll
        for (int mi = 0; mi < 2; ++mi) {
            bf16x16 a = ld_fragA(sA + (wm + mi * 16 + lrow) * LDK, lhi);
#pragma unroll
            for (int ni = 0; ni < 2; ++ni) {
                bf16x16 b = ld_fragB(sB + (wn + ni * 16 + lrow) * LDK, lhi);
                WMMA_BF16(acc[mi][ni], a, b);
            }
        }
        __syncthreads();
    }
#pragma unroll
    for (int mi = 0; mi < 2; ++mi)
#pragma unroll
        for (int ni = 0; ni < 2; ++ni)
#pragma unroll
            for (int r = 0; r < 8; ++r) {
                int m = m0 + wm + mi * 16 + r + (lhi ? 8 : 0);
                int n = n0 + wn + ni * 16 + lrow;
                P[m * SCH + n] = (bf16)acc[mi][ni][r];
            }
}

// ===========================================================================
// Kernel 2: attn[b,h,d,e] = sum_n Q[b,n,h,d] * K[b,n,h,e]   (K-dim = 8192)
// grid (2, 4, 16): x = e-tile(128), y = d-tile(64), z = b*8+h
// Transposed staging (element-wise, 2B) + speculative prefetch of next tile.
// ===========================================================================
__global__ void qtk_gemm(const bf16* __restrict__ Qb, const bf16* __restrict__ Kb,
                         float* __restrict__ attnF)
{
    const int bh = blockIdx.z, b = bh >> 3, h = bh & 7;
    const int d0 = blockIdx.y * 64;
    const int e0 = blockIdx.x * 128;

    __shared__ __align__(16) bf16 sA[64 * LDK];   // [d][n]
    __shared__ __align__(16) bf16 sB[128 * LDK];  // [e][n]

    const int tid  = threadIdx.x;
    const int wave = tid >> 5, lane = tid & 31;
    const int lrow = lane & 15, lhi = lane >> 4;
    const int wm = (wave >> 2) * 32;
    const int wn = (wave & 3) * 32;

    f32x8 acc[2][2] = {};

    for (int nt = 0; nt < SN; nt += 32) {
        {   // sA[d][n] = Q[b, nt+n, h, d0+d]   (contiguous d -> coalesced)
            const int d   = tid & 63;
            const int n0i = tid >> 6;   // 0..3
#pragma unroll
            for (int n = n0i; n < 32; n += 4)
                sA[d * LDK + n] = Qb[(b * SN + nt + n) * SCH + h * SC + d0 + d];
            if (nt + 32 < SN)   // speculative prefetch of next tile (L2 stream)
                __builtin_prefetch(&Qb[(b * SN + nt + 32 + n0i) * SCH + h * SC + d0 + d], 0, 0);
        }
        {   // sB[e][n] = K[b, nt+n, h, e0+e]
            const int e   = tid & 127;
            const int n0i = tid >> 7;   // 0..1
#pragma unroll
            for (int n = n0i; n < 32; n += 2)
                sB[e * LDK + n] = Kb[(b * SN + nt + n) * SCH + h * SC + e0 + e];
            if (nt + 32 < SN)
                __builtin_prefetch(&Kb[(b * SN + nt + 32 + n0i) * SCH + h * SC + e0 + e], 0, 0);
        }
        __syncthreads();
#pragma unroll
        for (int mi = 0; mi < 2; ++mi) {
            bf16x16 a = ld_fragA(sA + (wm + mi * 16 + lrow) * LDK, lhi);
#pragma unroll
            for (int ni = 0; ni < 2; ++ni) {
                bf16x16 bb = ld_fragB(sB + (wn + ni * 16 + lrow) * LDK, lhi);
                WMMA_BF16(acc[mi][ni], a, bb);
            }
        }
        __syncthreads();
    }
#pragma unroll
    for (int mi = 0; mi < 2; ++mi)
#pragma unroll
        for (int ni = 0; ni < 2; ++ni)
#pragma unroll
            for (int r = 0; r < 8; ++r) {
                int d = d0 + wm + mi * 16 + r + (lhi ? 8 : 0);
                int e = e0 + wn + ni * 16 + lrow;
                attnF[(bh * SC + d) * SC + e] = acc[mi][ni][r];
            }
}

// ===========================================================================
// Kernel 3: InstanceNorm2d (per b,h over 256x256, biased var) + softmax(axis e)
// grid (16), block 256.  Output bf16.
// ===========================================================================
__global__ void norm_softmax(const float* __restrict__ attnF, bf16* __restrict__ attnB)
{
    const int bh = blockIdx.x;
    const float* A = attnF + (size_t)bh * SC * SC;
    const int tid = threadIdx.x, wave = tid >> 5, lane = tid & 31;
    __shared__ float ws1[8], ws2[8];

    float s = 0.f, s2 = 0.f;
    for (int i = tid; i < SC * SC; i += 256) {
        float v = A[i];
        s += v; s2 += v * v;
    }
#pragma unroll
    for (int off = 16; off; off >>= 1) {
        s  += __shfl_xor(s,  off, 32);
        s2 += __shfl_xor(s2, off, 32);
    }
    if (lane == 0) { ws1[wave] = s; ws2[wave] = s2; }
    __syncthreads();
    if (wave == 0) {
        s  = (lane < 8) ? ws1[lane] : 0.f;
        s2 = (lane < 8) ? ws2[lane] : 0.f;
#pragma unroll
        for (int off = 4; off; off >>= 1) {
            s  += __shfl_xor(s,  off, 32);
            s2 += __shfl_xor(s2, off, 32);
        }
        if (lane == 0) { ws1[0] = s; ws2[0] = s2; }
    }
    __syncthreads();
    const float inv_n = 1.f / (float)(SC * SC);
    const float mu   = ws1[0] * inv_n;
    const float var  = ws2[0] * inv_n - mu * mu;
    const float rinv = rsqrtf(var + 1e-5f);

    // 8 waves, each wave handles rows wave, wave+8, ... ; lane covers 8 cols
    for (int row = wave; row < SC; row += 8) {
        const float* R = A + row * SC;
        float x[8];
        float mx = -3.0e38f;
#pragma unroll
        for (int j = 0; j < 8; ++j) {
            x[j] = (R[lane + j * 32] - mu) * rinv;
            mx = fmaxf(mx, x[j]);
        }
#pragma unroll
        for (int off = 16; off; off >>= 1) mx = fmaxf(mx, __shfl_xor(mx, off, 32));
        float sum = 0.f;
#pragma unroll
        for (int j = 0; j < 8; ++j) { x[j] = __expf(x[j] - mx); sum += x[j]; }
#pragma unroll
        for (int off = 16; off; off >>= 1) sum += __shfl_xor(sum, off, 32);
        const float r = 1.f / sum;
        bf16* O = attnB + (size_t)bh * SC * SC + row * SC;
#pragma unroll
        for (int j = 0; j < 8; ++j) O[lane + j * 32] = (bf16)(x[j] * r);
    }
}

// ===========================================================================
// Kernel 4: ctx2[b, n, d*H+h] = sum_e V[b,n,h,e] * attn[b,h,d,e]
// grid (2, 128, 16): x = d-tile(128), y = n-tile(64), z = b*8+h
// Both staging tiles are pure bf16 copies -> async global->LDS b128 path.
// ===========================================================================
__global__ void ctx_gemm(const bf16* __restrict__ Vb, const bf16* __restrict__ attnB,
                         bf16* __restrict__ ctx2)
{
    const int bh = blockIdx.z, b = bh >> 3, h = bh & 7;
    const int n0 = blockIdx.y * 64;
    const int d0 = blockIdx.x * 128;

    __shared__ __align__(16) bf16 sA[64 * LDK];   // [n][e]
    __shared__ __align__(16) bf16 sB[128 * LDK];  // [d][e]  (attn rows copied directly)

    const int tid  = threadIdx.x;
    const int wave = tid >> 5, lane = tid & 31;
    const int lrow = lane & 15, lhi = lane >> 4;
    const int wm = (wave >> 2) * 32;
    const int wn = (wave & 3) * 32;

    f32x8 acc[2][2] = {};

    for (int et = 0; et < SC; et += 32) {
#if HAVE_ASYNC
        {   // sA: 64 rows x 64B -> 256 b128 async transfers, 1/thread
            const int r = tid >> 2;           // row 0..63
            const int c = (tid & 3) * 8;      // 16B chunk (elements)
            ASYNC_CP16(sA + r * LDK + c,
                       Vb + (b * SN + n0 + r) * SCH + h * SC + et + c);
        }
        {   // sB: 128 rows x 64B -> 512 b128 async transfers, 2/thread
            const int r = tid >> 2;
            const int c = (tid & 3) * 8;
            ASYNC_CP16(sB + r * LDK + c,
                       attnB + (size_t)bh * SC * SC + (d0 + r) * SC + et + c);
            ASYNC_CP16(sB + (r + 64) * LDK + c,
                       attnB + (size_t)bh * SC * SC + (d0 + r + 64) * SC + et + c);
        }
        wait_async0();
#else
        {   // sA[n][e] = V[b, n0+n, h, et+e]
            const int e = tid & 31;
            const int r = tid >> 5;   // 0..7
#pragma unroll
            for (int rr = r; rr < 64; rr += 8)
                sA[rr * LDK + e] = Vb[(b * SN + n0 + rr) * SCH + h * SC + et + e];
        }
        {   // sB[d][e] = attn[bh, d0+d, et+e]
            const int e = tid & 31;
            const int d = tid >> 5;
#pragma unroll
            for (int dd = d; dd < 128; dd += 8)
                sB[dd * LDK + e] = attnB[(size_t)bh * SC * SC + (d0 + dd) * SC + et + e];
        }
#endif
        __syncthreads();
#pragma unroll
        for (int mi = 0; mi < 2; ++mi) {
            bf16x16 a = ld_fragA(sA + (wm + mi * 16 + lrow) * LDK, lhi);
#pragma unroll
            for (int ni = 0; ni < 2; ++ni) {
                bf16x16 bb = ld_fragB(sB + (wn + ni * 16 + lrow) * LDK, lhi);
                WMMA_BF16(acc[mi][ni], a, bb);
            }
        }
        __syncthreads();
    }
    // scatter: [B,N,Dh,H] flatten -> col index d*H + h
#pragma unroll
    for (int mi = 0; mi < 2; ++mi)
#pragma unroll
        for (int ni = 0; ni < 2; ++ni)
#pragma unroll
            for (int r = 0; r < 8; ++r) {
                int n = n0 + wm + mi * 16 + r + (lhi ? 8 : 0);
                int d = d0 + wn + ni * 16 + lrow;
                ctx2[(b * SN + n) * SCH + d * SH + h] = (bf16)acc[mi][ni][r];
            }
}

// ===========================================================================
// Kernel 5: out = ctx2[16384,2048] x Wo[2048,256]  -> fp32
// grid (2, 256).  A tile is a pure bf16 copy -> async path; B is transposed
// fp32->bf16 (stays element-wise) + prefetch.
// ===========================================================================
__global__ void out_gemm(const bf16* __restrict__ ctx2, const float* __restrict__ Wo,
                         float* __restrict__ out)
{
    const int m0 = blockIdx.y * 64;
    const int n0 = blockIdx.x * 128;

    __shared__ __align__(16) bf16 sA[64 * LDK];
    __shared__ __align__(16) bf16 sB[128 * LDK];

    const int tid  = threadIdx.x;
    const int wave = tid >> 5, lane = tid & 31;
    const int lrow = lane & 15, lhi = lane >> 4;
    const int wm = (wave >> 2) * 32;
    const int wn = (wave & 3) * 32;

    f32x8 acc[2][2] = {};

    for (int kt = 0; kt < SCH; kt += 32) {
#if HAVE_ASYNC
        {   // sA: 64 rows x 64B -> 256 b128 async transfers, 1/thread
            const int r = tid >> 2;
            const int c = (tid & 3) * 8;
            ASYNC_CP16(sA + r * LDK + c, ctx2 + (m0 + r) * SCH + kt + c);
        }
#else
        {   // sA[m][k] = ctx2[m0+m, kt+k]
            const int k = tid & 31;
            const int r = tid >> 5;
#pragma unroll
            for (int rr = r; rr < 64; rr += 8)
                sA[rr * LDK + k] = ctx2[(m0 + rr) * SCH + kt + k];
        }
#endif
        {   // sB[n][k] = (bf16)Wo[kt+k, n0+n]
            const int n  = tid & 127;
            const int k0 = tid >> 7;
#pragma unroll
            for (int kk = k0; kk < 32; kk += 2)
                sB[n * LDK + kk] = (bf16)Wo[(kt + kk) * SC + n0 + n];
            if (kt + 32 < SCH)
                __builtin_prefetch(&Wo[(kt + 32 + k0) * SC + n0 + n], 0, 0);
        }
#if HAVE_ASYNC
        wait_async0();
#endif
        __syncthreads();
#pragma unroll
        for (int mi = 0; mi < 2; ++mi) {
            bf16x16 a = ld_fragA(sA + (wm + mi * 16 + lrow) * LDK, lhi);
#pragma unroll
            for (int ni = 0; ni < 2; ++ni) {
                bf16x16 bb = ld_fragB(sB + (wn + ni * 16 + lrow) * LDK, lhi);
                WMMA_BF16(acc[mi][ni], a, bb);
            }
        }
        __syncthreads();
    }
#pragma unroll
    for (int mi = 0; mi < 2; ++mi)
#pragma unroll
        for (int ni = 0; ni < 2; ++ni)
#pragma unroll
            for (int r = 0; r < 8; ++r) {
                int m = m0 + wm + mi * 16 + r + (lhi ? 8 : 0);
                int n = n0 + wn + ni * 16 + lrow;
                out[m * SC + n] = acc[mi][ni][r];
            }
}

// ===========================================================================
extern "C" void kernel_launch(void* const* d_in, const int* in_sizes, int n_in,
                              void* d_out, int out_size, void* d_ws, size_t ws_size,
                              hipStream_t stream) {
    (void)in_sizes; (void)n_in; (void)out_size; (void)ws_size;
    const float* emb = (const float*)d_in[0];
    const float* Wq  = (const float*)d_in[1];
    const float* Wk  = (const float*)d_in[2];
    const float* Wv  = (const float*)d_in[3];
    const float* Wo  = (const float*)d_in[4];
    float* out = (float*)d_out;

    char* w = (char*)d_ws;
    const size_t MB = (size_t)1 << 20;
    bf16*  Qb    = (bf16*) (w);              //  64 MB  [16384,2048] bf16
    bf16*  Kb    = (bf16*) (w + 64  * MB);   //  64 MB
    bf16*  Vb    = (bf16*) (w + 128 * MB);   //  64 MB
    float* attnF = (float*)(w + 192 * MB);   //   4 MB  [16,256,256] fp32
    bf16*  attnB = (bf16*) (w + 196 * MB);   //   2 MB  [16,256,256] bf16
    bf16*  ctx2  = (bf16*) (w + 198 * MB);   //  64 MB  [16384,2048] bf16

    qkv_gemm   <<<dim3(SCH / 128, SM / 64, 3), 256, 0, stream>>>(emb, Wq, Wk, Wv, Qb, Kb, Vb);
    qtk_gemm   <<<dim3(SC / 128, SC / 64, SB * SH), 256, 0, stream>>>(Qb, Kb, attnF);
    norm_softmax<<<dim3(SB * SH), 256, 0, stream>>>(attnF, attnB);
    ctx_gemm   <<<dim3(SC / 128, SN / 64, SB * SH), 256, 0, stream>>>(Vb, attnB, ctx2);
    out_gemm   <<<dim3(SC / 128, SM / 64), 256, 0, stream>>>(ctx2, Wo, out);
}
